// EncDecAttention_39049842655851
// MI455X (gfx1250) — matrix operand order repaired and checked by reading, not compiled, not run on hardware
//
#include <hip/hip_runtime.h>

// ---------------------------------------------------------------------------
// EncDecAttention for MI455X (gfx1250): bf16 WMMA GEMMs + f32 softmax.
// Outputs concatenated in d_out: q, k, v, beta, scores, o (all f32).
// 128x128 block tile, 8 waves (32x64 per wave), double-buffered LDS pipeline.
// All LDS traffic is b128: fragment reads AND staging stores.
// ---------------------------------------------------------------------------

typedef __attribute__((ext_vector_type(16))) __bf16 v16bf;
typedef __attribute__((ext_vector_type(8)))  float  v8f;

union Frag {
    unsigned int u[8];
    v16bf v;
};

// round-to-nearest-even f32 -> bf16 pair packed into one dword (lo = first)
__device__ __forceinline__ unsigned int pack_bf16(float a, float b) {
    unsigned int au = __float_as_uint(a);
    unsigned int bu = __float_as_uint(b);
    au += 0x7FFFu + ((au >> 16) & 1u);
    bu += 0x7FFFu + ((bu >> 16) & 1u);
    return __builtin_amdgcn_perm(bu, au, 0x07060302u);  // {bu[31:16], au[31:16]}
}

// ---------------------------------------------------------------------------
// Generic tiled GEMM:  C[M,N] = op( A[M,K] * B )  with
//   BT=false : B is [K,N] row-major            (projections, o = scores @ v)
//   BT=true  : B is [N,K] row-major (B^T GEMM) (beta = q @ k^T)
//   BIASRELU : C = relu(acc + bias[col])
//   MASK     : C = acc * mask[row,col]
// Block = 256 threads (8 waves). Tile 128x128, K-step 32 (bf16 WMMA K).
// LDS tiles are stored in contiguous K order (lane half h gets k = 16h + s) so each
// lane's 16 bf16 operand values are one contiguous 32-byte run (ds_load_b128).
// ---------------------------------------------------------------------------
template <bool BT, bool BIASRELU, bool MASK>
__global__ __launch_bounds__(256) void gemm_bf16_wmma(
    const float* __restrict__ A, const float* __restrict__ Bm,
    const float* __restrict__ bias, const float* __restrict__ mask,
    float* __restrict__ C, int M, int N, int K, int lda, int ldb, int ldc,
    long long strideA, long long strideB, long long strideC,
    long long strideMask) {
    (void)M; (void)N;
    // double-buffered: [buf][128 rows/cols][16 packed K-pairs]
    __shared__ unsigned int ldsA[2 * 128 * 16];
    __shared__ unsigned int ldsB[2 * 128 * 16];

    const int t     = threadIdx.x;
    const int bz    = blockIdx.z;
    const int tileM = blockIdx.y * 128;
    const int tileN = blockIdx.x * 128;

    const float* Ab = A + (size_t)bz * strideA;
    const float* Bb = Bm + (size_t)bz * strideB;
    float*       Cb = C + (size_t)bz * strideC;

    const int wave = t >> 5;     // 0..7
    const int wr   = wave >> 1;  // row super-tile: rows [32*wr, 32*wr+32)
    const int wc   = wave & 1;   // col super-tile: cols [64*wc, 64*wc+64)
    const int lane = t & 31;
    const int lrow = lane & 15;
    const int kg   = lane >> 4;  // K-group select (ISA 16-bit operand layout)

    const v8f vzero = {0.f, 0.f, 0.f, 0.f, 0.f, 0.f, 0.f, 0.f};
    v8f acc[2][4];
#pragma unroll
    for (int mi = 0; mi < 2; ++mi)
#pragma unroll
        for (int c = 0; c < 4; ++c) acc[mi][c] = vzero;

    const int ksteps = K >> 5;

    // staged global data (registers) for the software pipeline
    float2 ra[8];
    float2 rb[8];

    // A-tile staging geometry: thread covers row (t>>1), K-halves by (t&1)
    const int arow_  = t >> 1;   // 0..127
    const int ahalf_ = t & 1;
    // B-tile staging geometry
    const int bcol_  = BT ? (t >> 1) : (t & 127);
    const int bhalf_ = BT ? (t & 1) : (t >> 7);

    // issue global loads for K-step kt (no wait here)
    auto global_fetch = [&](int kt) {
        const int kbase = kt << 5;
        {
            const float* src =
                Ab + (size_t)(tileM + arow_) * lda + kbase + ahalf_ * 16;
            if (kt + 2 < ksteps) __builtin_prefetch(src + 64, 0, 1);
#pragma unroll
            for (int i = 0; i < 8; ++i)
                ra[i] = *(const float2*)(src + 2 * i);
        }
        if (BT) {
            const float* src =
                Bb + (size_t)(tileN + bcol_) * ldb + kbase + bhalf_ * 16;
            if (kt + 2 < ksteps) __builtin_prefetch(src + 64, 0, 1);
#pragma unroll
            for (int i = 0; i < 8; ++i)
                rb[i] = *(const float2*)(src + 2 * i);
        } else {
            // B row-major [K,N]: thread owns K-pairs p = bhalf_*8 + i of its
            // column; loads coalesce across the 128 columns per K-row.
#pragma unroll
            for (int i = 0; i < 8; ++i) {
                const int p = bhalf_ * 8 + i;
                const float* s0 =
                    Bb + (size_t)(kbase + 2 * p) * ldb + tileN + bcol_;
                if (i == 0 && kt + 2 < ksteps)
                    __builtin_prefetch(s0 + (size_t)64 * ldb, 0, 1);
                rb[i].x = s0[0];
                rb[i].y = s0[ldb];
            }
        }
    };

    // pack staged registers and store into LDS buffer (kt & 1), b128 stores
    auto lds_store = [&](int kt) {
        const int buf = (kt & 1) * 2048;
        {
            // fragment layout: p in [half*8, half*8+8) maps to one contiguous
            // 8-dword run at s = half*8 (contiguous K order, matches B)
            unsigned int* dst = ldsA + buf + arow_ * 16;
            uint4 w0, w1;
            w0.x = pack_bf16(ra[0].x, ra[0].y);
            w0.y = pack_bf16(ra[1].x, ra[1].y);
            w0.z = pack_bf16(ra[2].x, ra[2].y);
            w0.w = pack_bf16(ra[3].x, ra[3].y);
            w1.x = pack_bf16(ra[4].x, ra[4].y);
            w1.y = pack_bf16(ra[5].x, ra[5].y);
            w1.z = pack_bf16(ra[6].x, ra[6].y);
            w1.w = pack_bf16(ra[7].x, ra[7].y);
            *(uint4*)(dst + ahalf_ * 8)     = w0;
            *(uint4*)(dst + ahalf_ * 8 + 4) = w1;
        }
        {
            // B: thread's 8 packed dwords are contiguous in its column
            unsigned int* dst = ldsB + buf + bcol_ * 16 + bhalf_ * 8;
            uint4 w0, w1;
            w0.x = pack_bf16(rb[0].x, rb[0].y);
            w0.y = pack_bf16(rb[1].x, rb[1].y);
            w0.z = pack_bf16(rb[2].x, rb[2].y);
            w0.w = pack_bf16(rb[3].x, rb[3].y);
            w1.x = pack_bf16(rb[4].x, rb[4].y);
            w1.y = pack_bf16(rb[5].x, rb[5].y);
            w1.z = pack_bf16(rb[6].x, rb[6].y);
            w1.w = pack_bf16(rb[7].x, rb[7].y);
            *(uint4*)(dst)     = w0;
            *(uint4*)(dst + 4) = w1;
        }
    };

    global_fetch(0);
    for (int kt = 0; kt < ksteps; ++kt) {
        lds_store(kt);
        __syncthreads();
        if (kt + 1 < ksteps) global_fetch(kt + 1);  // overlaps with WMMAs below

        const unsigned int* bufA = ldsA + (kt & 1) * 2048;
        const unsigned int* bufB = ldsB + (kt & 1) * 2048;

        Frag a[2];
#pragma unroll
        for (int mi = 0; mi < 2; ++mi) {
            const unsigned int* arow =
                bufA + (wr * 32 + mi * 16 + lrow) * 16 + kg * 8;
#pragma unroll
            for (int j = 0; j < 8; ++j) a[mi].u[j] = arow[j];
        }
#pragma unroll
        for (int c = 0; c < 4; ++c) {
            Frag b;
            const unsigned int* brow =
                bufB + (wc * 64 + c * 16 + lrow) * 16 + kg * 8;
#pragma unroll
            for (int j = 0; j < 8; ++j) b.u[j] = brow[j];
            acc[0][c] = __builtin_amdgcn_wmma_f32_16x16x32_bf16(
                false, a[0].v, false, b.v, (short)0, acc[0][c], false, false);
            acc[1][c] = __builtin_amdgcn_wmma_f32_16x16x32_bf16(
                false, a[1].v, false, b.v, (short)0, acc[1][c], false, false);
        }
        __syncthreads();
    }

    // ---- epilogue: C/D layout = lane[0:15] -> M=r, lane[16:31] -> M=8+r ----
#pragma unroll
    for (int mi = 0; mi < 2; ++mi) {
#pragma unroll
        for (int c = 0; c < 4; ++c) {
            const int col = tileN + wc * 64 + c * 16 + lrow;
            float bval = 0.f;
            if (BIASRELU) bval = bias[col];
#pragma unroll
            for (int r = 0; r < 8; ++r) {
                const int row = tileM + wr * 32 + mi * 16 + kg * 8 + r;
                float val = acc[mi][c][r];
                if (BIASRELU) val = fmaxf(val + bval, 0.f);
                if (MASK)
                    val *= mask[(size_t)bz * strideMask +
                                (size_t)row * ldc + col];
                Cb[(size_t)row * ldc + col] = val;
            }
        }
    }
}

// ---------------------------------------------------------------------------
// Row softmax over LK=2048: one block (256 thr, 8 waves) per row, 8 elems/thr.
// ---------------------------------------------------------------------------
__global__ __launch_bounds__(256) void softmax_rows(
    const float* __restrict__ beta, float* __restrict__ scores, int LK) {
    __shared__ float red[8];
    const int row = blockIdx.x;
    const int t   = threadIdx.x;
    const float* src = beta + (size_t)row * LK;
    float*       dst = scores + (size_t)row * LK;

    float v[8];
    const float4 a = *(const float4*)(src + t * 8);
    const float4 b = *(const float4*)(src + t * 8 + 4);
    v[0] = a.x; v[1] = a.y; v[2] = a.z; v[3] = a.w;
    v[4] = b.x; v[5] = b.y; v[6] = b.z; v[7] = b.w;

    float m = v[0];
#pragma unroll
    for (int j = 1; j < 8; ++j) m = fmaxf(m, v[j]);
#pragma unroll
    for (int off = 16; off > 0; off >>= 1) m = fmaxf(m, __shfl_xor(m, off, 32));
    if ((t & 31) == 0) red[t >> 5] = m;
    __syncthreads();
    m = red[0];
#pragma unroll
    for (int i = 1; i < 8; ++i) m = fmaxf(m, red[i]);
    __syncthreads();

    float s = 0.f;
#pragma unroll
    for (int j = 0; j < 8; ++j) {
        v[j] = __expf(v[j] - m);
        s += v[j];
    }
#pragma unroll
    for (int off = 16; off > 0; off >>= 1) s += __shfl_xor(s, off, 32);
    if ((t & 31) == 0) red[t >> 5] = s;
    __syncthreads();
    s = 0.f;
#pragma unroll
    for (int i = 0; i < 8; ++i) s += red[i];
    const float inv = 1.0f / s;

    float4 o1 = {v[0] * inv, v[1] * inv, v[2] * inv, v[3] * inv};
    float4 o2 = {v[4] * inv, v[5] * inv, v[6] * inv, v[7] * inv};
    *(float4*)(dst + t * 8) = o1;
    *(float4*)(dst + t * 8 + 4) = o2;
}

// ---------------------------------------------------------------------------
extern "C" void kernel_launch(void* const* d_in, const int* in_sizes, int n_in,
                              void* d_out, int out_size, void* d_ws,
                              size_t ws_size, hipStream_t stream) {
    (void)in_sizes; (void)n_in; (void)out_size; (void)d_ws; (void)ws_size;
    constexpr int B = 4, LQ = 2048, LK = 2048;
    constexpr int ENC = 1024, DEC = 1024, DQK = 128, DV = 1024;

    const float* dec   = (const float*)d_in[0];
    const float* enc   = (const float*)d_in[1];
    const float* masks = (const float*)d_in[2];
    const float* Wq    = (const float*)d_in[3];
    const float* bq    = (const float*)d_in[4];
    const float* Wk    = (const float*)d_in[5];
    const float* bk    = (const float*)d_in[6];
    const float* Wv    = (const float*)d_in[7];
    const float* bv    = (const float*)d_in[8];

    float* out    = (float*)d_out;
    float* q      = out;
    float* k      = q + (size_t)B * LQ * DQK;
    float* v      = k + (size_t)B * LK * DQK;
    float* beta   = v + (size_t)B * LK * DV;
    float* scores = beta + (size_t)B * LQ * LK;
    float* o      = scores + (size_t)B * LQ * LK;

    const dim3 blk(256);

    // q = relu(dec @ Wq + bq)   [8192,1024] x [1024,128]
    gemm_bf16_wmma<false, true, false>
        <<<dim3(DQK / 128, (B * LQ) / 128, 1), blk, 0, stream>>>(
            dec, Wq, bq, nullptr, q, B * LQ, DQK, DEC, DEC, DQK, DQK, 0, 0, 0, 0);
    // k = relu(enc @ Wk + bk)
    gemm_bf16_wmma<false, true, false>
        <<<dim3(DQK / 128, (B * LK) / 128, 1), blk, 0, stream>>>(
            enc, Wk, bk, nullptr, k, B * LK, DQK, ENC, ENC, DQK, DQK, 0, 0, 0, 0);
    // v = relu(enc @ Wv + bv)   [8192,1024] x [1024,1024]
    gemm_bf16_wmma<false, true, false>
        <<<dim3(DV / 128, (B * LK) / 128, 1), blk, 0, stream>>>(
            enc, Wv, bv, nullptr, v, B * LK, DV, ENC, ENC, DV, DV, 0, 0, 0, 0);
    // beta = (q @ k^T) * mask   per batch: [2048,128] x [2048,128]^T
    gemm_bf16_wmma<true, false, true>
        <<<dim3(LK / 128, LQ / 128, B), blk, 0, stream>>>(
            q, k, nullptr, masks, beta, LQ, LK, DQK, DQK, DQK, LK,
            (long long)LQ * DQK, (long long)LK * DQK, (long long)LQ * LK,
            (long long)LQ * LK);
    // scores = softmax(beta)
    softmax_rows<<<B * LQ, 256, 0, stream>>>(beta, scores, LK);
    // o = scores @ v            per batch: [2048,2048] x [2048,1024]
    gemm_bf16_wmma<false, false, false>
        <<<dim3(DV / 128, LQ / 128, B), blk, 0, stream>>>(
            scores, v, nullptr, nullptr, o, LQ, DV, LK, LK, DV, DV,
            (long long)LQ * LK, (long long)LK * DV, (long long)LQ * DV, 0);
}
